// kernel_generated_1_21887153341294
// MI455X (gfx1250) — matrix operand order repaired and checked by reading, not compiled, not run on hardware
//
#include <hip/hip_runtime.h>

typedef __attribute__((ext_vector_type(16))) __bf16 v16bf;
typedef __attribute__((ext_vector_type(8)))  __bf16 v8bf;
typedef __attribute__((ext_vector_type(8)))  float  v8f;

#define CIN_   128
#define COUT_  32
#define H_     28
#define W_     28
#define KW_    3
#define KTOT_  (CIN_ * KW_)        // 384
#define APITCH 392                 // weight K-pitch (384 + 8) -> conflict-free, 16B aligned frags
#define XPITCH 136                 // x channel-pitch (128 + 8) -> conflict-free, 16B aligned frags
#define XPOS   30                  // W + 2 halo positions
#define WAVES_ 4
#define SMEM_BYTES ((2 * COUT_ * APITCH + 2 * WAVES_ * XPOS * XPITCH) * 2)  // 115,456 B

// ---- bf16 helpers via bit ops (RNE), storage-only bf16 ----
static __device__ __forceinline__ unsigned short bf16_rne(float f) {
  unsigned int u = __float_as_uint(f);
  u += 0x7FFFu + ((u >> 16) & 1u);
  return (unsigned short)(u >> 16);
}
static __device__ __forceinline__ float bf16_to_f32(unsigned short h) {
  return __uint_as_float(((unsigned int)h) << 16);
}

static __device__ __forceinline__ v16bf frag_cat(v8bf a, v8bf b) {
  return __builtin_shufflevector(a, b, 0,1,2,3,4,5,6,7,8,9,10,11,12,13,14,15);
}
// A-fragment: 16 contiguous K per lane (matches B-fragment)
static __device__ __forceinline__ v16bf ld_fragA(const unsigned short* p) {
  return frag_cat(*(const v8bf*)p, *(const v8bf*)(p + 8));
}
// B-fragment: 16 contiguous K (channels) per lane
static __device__ __forceinline__ v16bf ld_fragB(const unsigned short* p) {
  return frag_cat(*(const v8bf*)p, *(const v8bf*)(p + 8));
}
static __device__ __forceinline__ v8f wmma_bf16(v16bf a, v16bf b, v8f c) {
  return __builtin_amdgcn_wmma_f32_16x16x32_bf16(false, a, false, b, (short)0, c,
                                                 false, false);
}
static __device__ __forceinline__ void store_rows(float* __restrict__ o, size_t base,
                                                  int jb, int n, v8f c) {
  if (n >= W_) return;
  #pragma unroll
  for (int r = 0; r < 8; ++r)
    o[base + (size_t)(jb + r) * (H_ * W_) + n] = c[r];
}

__global__ void __launch_bounds__(WAVES_ * 32)
conv1dw_roll_wmma(const float* __restrict__ x, const float* __restrict__ w,
                  float* __restrict__ out) {
  extern __shared__ __align__(16) char smem_raw[];
  unsigned short* sAh = (unsigned short*)smem_raw;       // [COUT][APITCH], kidx = k*128+i
  unsigned short* sAl = sAh + COUT_ * APITCH;
  unsigned short* sXh = sAl + COUT_ * APITCH;            // [WAVES][XPOS][XPITCH]
  unsigned short* sXl = sXh + WAVES_ * XPOS * XPITCH;

  const int tid  = threadIdx.x;
  const int wave = tid >> 5;
  const int lane = tid & 31;

  // ---- stage weights, hi/lo bf16 split, K-order kidx = k*CIN + i ----
  for (int t = tid; t < COUT_ * KTOT_; t += WAVES_ * 32) {
    int m = t / KTOT_;
    int r = t - m * KTOT_;
    int i = r / KW_;
    int k = r - i * KW_;
    float f = w[t];                       // w is (COUT, CIN, KW) row-major
    unsigned short hi = bf16_rne(f);
    unsigned short lo = bf16_rne(f - bf16_to_f32(hi));
    int idx = m * APITCH + k * CIN_ + i;
    sAh[idx] = hi;
    sAl[idx] = lo;
  }

  // ---- stage this wave's (b,h) row with W-halo, layout [pos][chan] ----
  const int row = blockIdx.x * WAVES_ + wave;
  const int b   = row / H_;
  const int h   = row - b * H_;
  unsigned short* xh = sXh + wave * (XPOS * XPITCH);
  unsigned short* xl = sXl + wave * (XPOS * XPITCH);
  for (int i = lane; i < CIN_; i += 32) {                // zero halos p=0, p=29
    xh[i] = 0; xl[i] = 0;
    xh[(XPOS - 1) * XPITCH + i] = 0;
    xl[(XPOS - 1) * XPITCH + i] = 0;
  }
  const float* xrow = x + ((size_t)b * CIN_ * H_ + h) * W_;
  for (int t = lane; t < CIN_ * W_; t += 32) {
    int i  = t / W_;
    int wp = t - i * W_;
    float f = xrow[(size_t)i * (H_ * W_) + wp];
    unsigned short hi = bf16_rne(f);
    unsigned short lo = bf16_rne(f - bf16_to_f32(hi));
    int idx = (wp + 1) * XPITCH + i;
    xh[idx] = hi;
    xl[idx] = lo;
  }
  __syncthreads();

  // ---- implicit GEMM: D(32 x 32cols) += A(32x384) * B(384x32cols) ----
  const int lm = lane & 15;
  const int kh = lane >> 4;
  v8f acc00 = {0.f,0.f,0.f,0.f,0.f,0.f,0.f,0.f};
  v8f acc01 = {0.f,0.f,0.f,0.f,0.f,0.f,0.f,0.f};
  v8f acc10 = {0.f,0.f,0.f,0.f,0.f,0.f,0.f,0.f};
  v8f acc11 = {0.f,0.f,0.f,0.f,0.f,0.f,0.f,0.f};
  const int n1c = (lm + 16 > W_ - 1) ? (W_ - 1) : (lm + 16);  // clamp garbage cols in-bounds

  for (int kk = 0; kk < 12; ++kk) {          // 12 K-steps of 32: k = kk>>2, i = (kk&3)*32 + ...
    const int ktap  = kk >> 2;
    const int ibase = (kk & 3) << 5;
    const int arow  = kk * 32 + kh * 16;
    v16bf A0h = ld_fragA(sAh + lm * APITCH + arow);
    v16bf A0l = ld_fragA(sAl + lm * APITCH + arow);
    v16bf A1h = ld_fragA(sAh + (lm + 16) * APITCH + arow);
    v16bf A1l = ld_fragA(sAl + (lm + 16) * APITCH + arow);
    const int b0 = (lm  + ktap) * XPITCH + ibase + kh * 16;
    const int b1 = (n1c + ktap) * XPITCH + ibase + kh * 16;
    v16bf B0h = ld_fragB(xh + b0);
    v16bf B0l = ld_fragB(xl + b0);
    v16bf B1h = ld_fragB(xh + b1);
    v16bf B1l = ld_fragB(xl + b1);
    // fp32-faithful: Ah*Bh + Ah*Bl + Al*Bh (Al*Bl term ~2^-18, dropped)
    acc00 = wmma_bf16(A0h, B0h, acc00);
    acc00 = wmma_bf16(A0h, B0l, acc00);
    acc00 = wmma_bf16(A0l, B0h, acc00);
    acc01 = wmma_bf16(A0h, B1h, acc01);
    acc01 = wmma_bf16(A0h, B1l, acc01);
    acc01 = wmma_bf16(A0l, B1h, acc01);
    acc10 = wmma_bf16(A1h, B0h, acc10);
    acc10 = wmma_bf16(A1h, B0l, acc10);
    acc10 = wmma_bf16(A1l, B0h, acc10);
    acc11 = wmma_bf16(A1h, B1h, acc11);
    acc11 = wmma_bf16(A1h, B1l, acc11);
    acc11 = wmma_bf16(A1l, B1h, acc11);
  }

  // ---- epilogue: roll(+1) along H folded into the store address ----
  const int hout = (h == H_ - 1) ? 0 : h + 1;
  const size_t obase = (size_t)b * COUT_ * H_ * W_ + (size_t)hout * W_;
  store_rows(out, obase,      kh * 8,      lm, acc00);   // M-tile 0, cols 0..15
  store_rows(out, obase,      kh * 8, 16 + lm, acc01);   // M-tile 0, cols 16..27
  store_rows(out, obase, 16 + kh * 8,      lm, acc10);   // M-tile 1, cols 0..15
  store_rows(out, obase, 16 + kh * 8, 16 + lm, acc11);   // M-tile 1, cols 16..27
}

extern "C" void kernel_launch(void* const* d_in, const int* in_sizes, int n_in,
                              void* d_out, int out_size, void* d_ws, size_t ws_size,
                              hipStream_t stream) {
  (void)in_sizes; (void)n_in; (void)out_size; (void)d_ws; (void)ws_size;
  const float* x = (const float*)d_in[0];
  const float* w = (const float*)d_in[1];
  float* out = (float*)d_out;

  (void)hipFuncSetAttribute(reinterpret_cast<const void*>(&conv1dw_roll_wmma),
                            hipFuncAttributeMaxDynamicSharedMemorySize,
                            (int)SMEM_BYTES);

  const int rows   = 512 * H_;             // 14336 (b,h) rows
  const int blocks = rows / WAVES_;        // 3584
  conv1dw_roll_wmma<<<blocks, WAVES_ * 32, SMEM_BYTES, stream>>>(x, w, out);
}
